// CausalSelfAttention_7971459301941
// MI455X (gfx1250) — matrix operand order repaired and checked by reading, not compiled, not run on hardware
//
#include <hip/hip_runtime.h>
#include <stdint.h>

typedef __attribute__((ext_vector_type(16))) __bf16 v16bf;
typedef __attribute__((ext_vector_type(8)))  float  v8f;
typedef int vi4 __attribute__((vector_size(16)));   // matches builtin prototype (GCC vector)

#define DEVINL __device__ __forceinline__

#if defined(__has_builtin)
#if __has_builtin(__builtin_amdgcn_global_load_async_to_lds_b128) && \
    __has_builtin(__builtin_amdgcn_s_wait_asynccnt)
#define HAS_ASYNC_LDS 1
#endif
#endif
#ifndef HAS_ASYNC_LDS
#define HAS_ASYNC_LDS 0
#endif

constexpr int Bn = 4, Tn = 2048, Cn = 1024, Hn = 16, HD = 64;
constexpr int Mrows = Bn * Tn;      // 8192
constexpr int N_QKV = 3 * Cn;       // 3072

DEVINL unsigned short f2bf(float f) {
  unsigned int u = __float_as_uint(f);
  u += 0x7FFFu + ((u >> 16) & 1u);   // round to nearest even
  return (unsigned short)(u >> 16);
}

union Frag32 { uint4 u[2]; v16bf v; unsigned short s[16]; };

// A-matrix fragment (16x32 bf16): lane holds row M=lane&15,
// K(e) = hi*16 + e  -> two contiguous 16B chunks at +hi*16 and +hi*16+8
DEVINL v16bf ldA(const unsigned short* row, int hi) {
  Frag32 f;
  f.u[0] = *(const uint4*)(row + hi * 16);
  f.u[1] = *(const uint4*)(row + hi * 16 + 8);
  return f.v;
}
// B-matrix fragment (32x16 bf16): lane holds column N=lane&15, K(e)=hi*16+e
DEVINL v16bf ldB(const unsigned short* base) {
  Frag32 f;
  f.u[0] = *(const uint4*)(base);
  f.u[1] = *(const uint4*)(base + 8);
  return f.v;
}
// B fragment gathered from a raw [K][64] tile at column d (strided)
DEVINL v16bf ldBcol(const unsigned short* tile, int d, int hi) {
  Frag32 f;
#pragma unroll
  for (int e = 0; e < 16; e++) f.s[e] = tile[(hi * 16 + e) * HD + d];
  return f.v;
}

#if HAS_ASYNC_LDS
// 16B per lane, global -> LDS, tracked by ASYNCcnt (no VGPR data path)
DEVINL void async_cp16(const unsigned short* g, unsigned short* l) {
  __builtin_amdgcn_global_load_async_to_lds_b128(
      (__attribute__((address_space(1))) vi4*)(uintptr_t)g,
      (__attribute__((address_space(3))) vi4*)(uintptr_t)l, 0, 0);
}
#endif

DEVINL void sync_cp16x2(const unsigned short* g, unsigned short* l) {
  uint4 a0 = *(const uint4*)(g);
  uint4 a1 = *(const uint4*)(g + 8);
  *(uint4*)(l) = a0;
  *(uint4*)(l + 8) = a1;
}

__global__ void cvt_f32_bf16(const float* __restrict__ in, unsigned short* __restrict__ out, int n) {
  int i = blockIdx.x * blockDim.x + threadIdx.x;
  int stride = gridDim.x * blockDim.x;
  for (; i < n; i += stride) out[i] = f2bf(in[i]);
}

// ---------------- QKV projection: [8192,1024] @ [3072,1024]^T -> scatter bf16 Q/K/V [B,H,T,64]
__global__ __launch_bounds__(128) void gemm_qkv(const unsigned short* __restrict__ xb,
                                                const unsigned short* __restrict__ wab,
                                                unsigned short* __restrict__ qb,
                                                unsigned short* __restrict__ kb,
                                                unsigned short* __restrict__ vb) {
  __shared__ unsigned short Al[2][64 * 32];
  __shared__ unsigned short Bl[2][64 * 32];
  int bm = blockIdx.x % (Mrows / 64);
  int bn = blockIdx.x / (Mrows / 64);
  int m0 = bm * 64, n0 = bn * 64;
  int tid = threadIdx.x;
  int lane = tid & 31, wave = tid >> 5;
  int col = lane & 15, hi = lane >> 4;
  int wm = (wave >> 1) * 32, wn = (wave & 1) * 32;
  int srow = tid >> 1, sk = (tid & 1) * 16;
  v8f acc[2][2] = {};

  auto stageAB = [&](int buf, int k0) {
    const unsigned short* ga = xb + (size_t)(m0 + srow) * Cn + k0 + sk;
    const unsigned short* gb = wab + (size_t)(n0 + srow) * Cn + k0 + sk;
#if HAS_ASYNC_LDS
    async_cp16(ga,     &Al[buf][srow * 32 + sk]);
    async_cp16(ga + 8, &Al[buf][srow * 32 + sk + 8]);
    async_cp16(gb,     &Bl[buf][srow * 32 + sk]);
    async_cp16(gb + 8, &Bl[buf][srow * 32 + sk + 8]);
#else
    sync_cp16x2(ga, &Al[buf][srow * 32 + sk]);
    sync_cp16x2(gb, &Bl[buf][srow * 32 + sk]);
#endif
  };

  stageAB(0, 0);
#if HAS_ASYNC_LDS
  __builtin_amdgcn_s_wait_asynccnt(0);
#endif
  __syncthreads();

  for (int k0 = 0; k0 < Cn; k0 += 32) {
    int cur = (k0 >> 5) & 1;
    if (k0 + 32 < Cn) stageAB(cur ^ 1, k0 + 32);
    if (k0 + 64 < Cn) {
      __builtin_prefetch(xb + (size_t)(m0 + srow) * Cn + k0 + 64 + sk, 0, 1);
      __builtin_prefetch(wab + (size_t)(n0 + srow) * Cn + k0 + 64 + sk, 0, 1);
    }
    v16bf af[2], bf[2];
#pragma unroll
    for (int t = 0; t < 2; t++) af[t] = ldA(&Al[cur][(wm + t * 16 + col) * 32], hi);
#pragma unroll
    for (int t = 0; t < 2; t++) bf[t] = ldB(&Bl[cur][(wn + t * 16 + col) * 32 + hi * 16]);
#pragma unroll
    for (int ti = 0; ti < 2; ti++)
#pragma unroll
      for (int tj = 0; tj < 2; tj++)
        acc[ti][tj] = __builtin_amdgcn_wmma_f32_16x16x32_bf16(
            false, af[ti], false, bf[tj], (short)0, acc[ti][tj], false, false);
#if HAS_ASYNC_LDS
    __builtin_amdgcn_s_wait_asynccnt(0);
#endif
    __syncthreads();
  }

#pragma unroll
  for (int ti = 0; ti < 2; ti++)
#pragma unroll
    for (int tj = 0; tj < 2; tj++) {
#pragma unroll
      for (int i = 0; i < 8; i++) {
        int m = m0 + wm + ti * 16 + hi * 8 + i;
        int n = n0 + wn + tj * 16 + col;
        int b = m >> 11, t = m & (Tn - 1);
        int which = n >> 10, c = n & (Cn - 1);
        int h = c >> 6, d = c & 63;
        unsigned short* dst = which == 0 ? qb : (which == 1 ? kb : vb);
        dst[((size_t)(b * Hn + h) * Tn + t) * HD + d] = f2bf(acc[ti][tj][i]);
      }
    }
}

// ---------------- Flash attention: per block = one (b,h) x 64-query tile; 4 waves x 16 queries
__global__ __launch_bounds__(128) void attn(const unsigned short* __restrict__ qb,
                                            const unsigned short* __restrict__ kb,
                                            const unsigned short* __restrict__ vb,
                                            unsigned short* __restrict__ yb) {
  __shared__ unsigned short Kl[2][32 * 64];    // K tile, [key][d]
  __shared__ unsigned short Vl[2][32 * 64];    // V tile, [key][d] (raw; column-gathered for B-frags)
  __shared__ unsigned short Pl[4 * 16 * 32];   // per-wave P tiles
  int qtile = blockIdx.x & 31;                 // T/64 = 32 tiles
  int bh = blockIdx.x >> 5;                    // 0..63
  int b = bh >> 4, h = bh & 15;
  int q0 = qtile * 64;
  int tid = threadIdx.x, lane = tid & 31, wave = tid >> 5;
  int col = lane & 15, hi = lane >> 4;
  int qw0 = q0 + wave * 16;
  const unsigned short* qbh = qb + (size_t)bh * Tn * HD;
  const unsigned short* kbh = kb + (size_t)bh * Tn * HD;
  const unsigned short* vbh = vb + (size_t)bh * Tn * HD;

  v16bf qf[2];
#pragma unroll
  for (int s = 0; s < 2; s++)
    qf[s] = ldA(qbh + (size_t)(qw0 + col) * HD + 32 * s, hi);

  v8f acco[4] = {};
  float mrow[8], lrow[8];
#pragma unroll
  for (int i = 0; i < 8; i++) { mrow[i] = -1e30f; lrow[i] = 0.f; }

  int nkt = qtile * 2 + 2;                     // key tiles of 32 up to q0+64
  int srow = tid >> 2, sdc = (tid & 3) * 16;   // staging map: 32 rows x 64 cols
  unsigned short* pw = Pl + wave * 16 * 32;

  auto stageKV = [&](int buf, int kt0) {
    const unsigned short* gk = kbh + (size_t)(kt0 + srow) * HD + sdc;
    const unsigned short* gv = vbh + (size_t)(kt0 + srow) * HD + sdc;
#if HAS_ASYNC_LDS
    async_cp16(gk,     &Kl[buf][srow * 64 + sdc]);
    async_cp16(gk + 8, &Kl[buf][srow * 64 + sdc + 8]);
    async_cp16(gv,     &Vl[buf][srow * 64 + sdc]);
    async_cp16(gv + 8, &Vl[buf][srow * 64 + sdc + 8]);
#else
    sync_cp16x2(gk, &Kl[buf][srow * 64 + sdc]);
    sync_cp16x2(gv, &Vl[buf][srow * 64 + sdc]);
#endif
  };

  stageKV(0, 0);
#if HAS_ASYNC_LDS
  __builtin_amdgcn_s_wait_asynccnt(0);
#endif
  __syncthreads();

  for (int kt = 0; kt < nkt; kt++) {
    int kt0 = kt * 32;
    int cur = kt & 1;
    if (kt + 1 < nkt) stageKV(cur ^ 1, kt0 + 32);

    if (kt0 <= qw0 + 15) {                     // wave-uniform causal tile skip
      v8f sa[2] = {};
#pragma unroll
      for (int j = 0; j < 2; j++)
#pragma unroll
        for (int s = 0; s < 2; s++) {
          v16bf kf = ldB(&Kl[cur][(j * 16 + col) * 64 + s * 32 + hi * 16]);
          sa[j] = __builtin_amdgcn_wmma_f32_16x16x32_bf16(
              false, qf[s], false, kf, (short)0, sa[j], false, false);
        }
      float sv[2][8];
      bool needMask = (kt0 + 31 > qw0);
#pragma unroll
      for (int j = 0; j < 2; j++)
#pragma unroll
        for (int i = 0; i < 8; i++) {
          float x = sa[j][i] * 0.125f;         // 1/sqrt(64)
          if (needMask) {
            int key = kt0 + j * 16 + col;
            int qq = qw0 + hi * 8 + i;
            if (key > qq) x = -1e30f;
          }
          sv[j][i] = x;
        }
#pragma unroll
      for (int i = 0; i < 8; i++) {
        float t = fmaxf(sv[0][i], sv[1][i]);
#pragma unroll
        for (int m = 1; m < 16; m <<= 1) t = fmaxf(t, __shfl_xor(t, m, 32));
        float mn = fmaxf(mrow[i], t);
        float sc = __expf(mrow[i] - mn);
        float p0 = __expf(sv[0][i] - mn);
        float p1 = __expf(sv[1][i] - mn);
        float rs = p0 + p1;
#pragma unroll
        for (int m = 1; m < 16; m <<= 1) rs += __shfl_xor(rs, m, 32);
        lrow[i] = lrow[i] * sc + rs;
        mrow[i] = mn;
#pragma unroll
        for (int jo = 0; jo < 4; jo++) acco[jo][i] *= sc;
        pw[(hi * 8 + i) * 32 + col] = f2bf(p0);
        pw[(hi * 8 + i) * 32 + 16 + col] = f2bf(p1);
      }
      v16bf pf = ldA(pw + col * 32, hi);
#pragma unroll
      for (int jo = 0; jo < 4; jo++) {
        v16bf vf = ldBcol(&Vl[cur][0], jo * 16 + col, hi);
        acco[jo] = __builtin_amdgcn_wmma_f32_16x16x32_bf16(
            false, pf, false, vf, (short)0, acco[jo], false, false);
      }
    }
#if HAS_ASYNC_LDS
    __builtin_amdgcn_s_wait_asynccnt(0);
#endif
    __syncthreads();
  }

#pragma unroll
  for (int i = 0; i < 8; i++) {
    float inv = 1.0f / lrow[i];
    int trow = qw0 + hi * 8 + i;
    size_t base = ((size_t)(b * Tn) + trow) * Cn + h * HD;
#pragma unroll
    for (int jo = 0; jo < 4; jo++)
      yb[base + jo * 16 + col] = f2bf(acco[jo][i] * inv);
  }
}

// ---------------- Output projection: y[8192,1024] @ w_proj[1024,1024]^T -> f32 out
__global__ __launch_bounds__(128) void gemm_proj(const unsigned short* __restrict__ yb,
                                                 const unsigned short* __restrict__ wpb,
                                                 float* __restrict__ out) {
  __shared__ unsigned short Al[2][64 * 32];
  __shared__ unsigned short Bl[2][64 * 32];
  int bm = blockIdx.x % (Mrows / 64);
  int bn = blockIdx.x / (Mrows / 64);
  int m0 = bm * 64, n0 = bn * 64;
  int tid = threadIdx.x;
  int lane = tid & 31, wave = tid >> 5;
  int col = lane & 15, hi = lane >> 4;
  int wm = (wave >> 1) * 32, wn = (wave & 1) * 32;
  int srow = tid >> 1, sk = (tid & 1) * 16;
  v8f acc[2][2] = {};

  auto stageAB = [&](int buf, int k0) {
    const unsigned short* ga = yb + (size_t)(m0 + srow) * Cn + k0 + sk;
    const unsigned short* gb = wpb + (size_t)(n0 + srow) * Cn + k0 + sk;
#if HAS_ASYNC_LDS
    async_cp16(ga,     &Al[buf][srow * 32 + sk]);
    async_cp16(ga + 8, &Al[buf][srow * 32 + sk + 8]);
    async_cp16(gb,     &Bl[buf][srow * 32 + sk]);
    async_cp16(gb + 8, &Bl[buf][srow * 32 + sk + 8]);
#else
    sync_cp16x2(ga, &Al[buf][srow * 32 + sk]);
    sync_cp16x2(gb, &Bl[buf][srow * 32 + sk]);
#endif
  };

  stageAB(0, 0);
#if HAS_ASYNC_LDS
  __builtin_amdgcn_s_wait_asynccnt(0);
#endif
  __syncthreads();

  for (int k0 = 0; k0 < Cn; k0 += 32) {
    int cur = (k0 >> 5) & 1;
    if (k0 + 32 < Cn) stageAB(cur ^ 1, k0 + 32);
    if (k0 + 64 < Cn) {
      __builtin_prefetch(yb + (size_t)(m0 + srow) * Cn + k0 + 64 + sk, 0, 1);
      __builtin_prefetch(wpb + (size_t)(n0 + srow) * Cn + k0 + 64 + sk, 0, 1);
    }
    v16bf af[2], bf[2];
#pragma unroll
    for (int t = 0; t < 2; t++) af[t] = ldA(&Al[cur][(wm + t * 16 + col) * 32], hi);
#pragma unroll
    for (int t = 0; t < 2; t++) bf[t] = ldB(&Bl[cur][(wn + t * 16 + col) * 32 + hi * 16]);
#pragma unroll
    for (int ti = 0; ti < 2; ti++)
#pragma unroll
      for (int tj = 0; tj < 2; tj++)
        acc[ti][tj] = __builtin_amdgcn_wmma_f32_16x16x32_bf16(
            false, af[ti], false, bf[tj], (short)0, acc[ti][tj], false, false);
#if HAS_ASYNC_LDS
    __builtin_amdgcn_s_wait_asynccnt(0);
#endif
    __syncthreads();
  }

#pragma unroll
  for (int ti = 0; ti < 2; ti++)
#pragma unroll
    for (int tj = 0; tj < 2; tj++)
#pragma unroll
      for (int i = 0; i < 8; i++) {
        int m = m0 + wm + ti * 16 + hi * 8 + i;
        int n = n0 + wn + tj * 16 + col;
        out[(size_t)m * Cn + n] = acc[ti][tj][i];
      }
}

extern "C" void kernel_launch(void* const* d_in, const int* in_sizes, int n_in,
                              void* d_out, int out_size, void* d_ws, size_t ws_size,
                              hipStream_t stream) {
  (void)in_sizes; (void)n_in; (void)out_size; (void)ws_size;
  const float* x = (const float*)d_in[0];
  const float* w_attn = (const float*)d_in[1];
  const float* w_proj = (const float*)d_in[2];

  char* ws = (char*)d_ws;
  size_t off = 0;
  auto alloc = [&](size_t bytes) -> void* {
    void* p = ws + off;
    off += (bytes + 255) & ~(size_t)255;
    return p;
  };
  unsigned short* xb  = (unsigned short*)alloc((size_t)Mrows * Cn * 2);
  unsigned short* wab = (unsigned short*)alloc((size_t)N_QKV * Cn * 2);
  unsigned short* wpb = (unsigned short*)alloc((size_t)Cn * Cn * 2);
  unsigned short* qb  = (unsigned short*)alloc((size_t)Bn * Hn * Tn * HD * 2);
  unsigned short* kb  = (unsigned short*)alloc((size_t)Bn * Hn * Tn * HD * 2);
  unsigned short* vb  = (unsigned short*)alloc((size_t)Bn * Hn * Tn * HD * 2);
  unsigned short* yb  = (unsigned short*)alloc((size_t)Mrows * Cn * 2);

  int nx = Mrows * Cn, nwa = N_QKV * Cn, nwp = Cn * Cn;
  cvt_f32_bf16<<<(nx + 255) / 256, 256, 0, stream>>>(x, xb, nx);
  cvt_f32_bf16<<<(nwa + 255) / 256, 256, 0, stream>>>(w_attn, wab, nwa);
  cvt_f32_bf16<<<(nwp + 255) / 256, 256, 0, stream>>>(w_proj, wpb, nwp);

  gemm_qkv<<<(Mrows / 64) * (N_QKV / 64), 128, 0, stream>>>(xb, wab, qb, kb, vb);
  attn<<<Bn * Hn * (Tn / 64), 128, 0, stream>>>(qb, kb, vb, yb);
  gemm_proj<<<(Mrows / 64) * (Cn / 64), 128, 0, stream>>>(yb, wpb, (float*)d_out);
}
